// KirchhoffMigration_22058952032317
// MI455X (gfx1250) — matrix operand order repaired and checked by reading, not compiled, not run on hardware
//
#include <hip/hip_runtime.h>
#include <hip/hip_bf16.h>

typedef __attribute__((ext_vector_type(16))) __bf16        v16bf;
typedef __attribute__((ext_vector_type(8)))  float         v8f;
typedef __attribute__((ext_vector_type(8)))  unsigned int  v8u;

#define KFREQ 16
#define NTR   64          // T*R
#define NBATCH 32
#define KTR   1024        // KFREQ*NTR
#define NVOX  131072      // 64*64*32
#define WAVES_PER_BLOCK 4
#define VOX_PER_WAVE 16
#define VOX_PER_BLOCK (WAVES_PER_BLOCK * VOX_PER_WAVE)
#define KSTEPS 32         // KTR / 32
#define FRAG_ELEMS (KSTEPS * 2 * 32 * 16)   // 32768 bf16 per component

// sign-flip all 16 packed bf16 lanes of a fragment (cheap negate)
static __device__ __forceinline__ v16bf negbf(v16bf x) {
    v8u u = __builtin_bit_cast(v8u, x);
    #pragma unroll
    for (int i = 0; i < 8; ++i) u[i] ^= 0x80008000u;
    return __builtin_bit_cast(v16bf, u);
}

// ---------------------------------------------------------------------------
// Pre-pass: convert y (f32) into WMMA-A-operand fragment-linear bf16 layout.
// Fragment for (s, mt): per-lane 16 contiguous bf16 (32B) -> b128-pair loads.
// ---------------------------------------------------------------------------
__global__ void KirchhoffMigration_yfrag_kernel(const float* __restrict__ y_re,
                                                const float* __restrict__ y_im,
                                                __bf16* __restrict__ fre,
                                                __bf16* __restrict__ fim)
{
    const int idx = blockIdx.x * blockDim.x + threadIdx.x;   // 0..32767
    if (idx >= FRAG_ELEMS) return;
    const int j    = idx & 15;
    const int lane = (idx >> 4) & 31;
    const int mt   = (idx >> 9) & 1;
    const int s    = idx >> 10;
    const int n  = (lane & 15) + mt * 16;
    const int kb = (lane >> 4) << 3;          // 0 (lo lanes) / 8 (hi lanes)
    const int v  = j >> 1, p = j & 1;
    const int kl = ((v >> 2) << 4) + kb + ((v & 3) << 1) + p; // K-local 0..31
    const int src = n * KTR + s * 32 + kl;
    fre[idx] = (__bf16)y_re[src];
    fim[idx] = (__bf16)y_im[src];
}

// ---------------------------------------------------------------------------
// Main fused kernel: geometry -> LDS, physics A on the fly, bf16 WMMA GEMM.
// ---------------------------------------------------------------------------
__global__ void __launch_bounds__(128)
KirchhoffMigration_22058952032317_kernel(
    const float* __restrict__ freqs,      // (16)
    const float* __restrict__ txp,        // (8,3)
    const float* __restrict__ rxp,        // (8,3)
    const float* __restrict__ xc,         // (64)
    const float* __restrict__ yc,         // (64)
    const float* __restrict__ zc,         // (32)
    const v16bf* __restrict__ yfre,       // fragment-linear bf16 y_re
    const v16bf* __restrict__ yfim,       // fragment-linear bf16 y_im
    float* __restrict__ out,              // (32, 131072) magnitudes
    unsigned int* __restrict__ gmax)      // float-bit max (mags >= 0)
{
    __shared__ float sRsum [WAVES_PER_BLOCK][NTR][VOX_PER_WAVE + 1];
    __shared__ float sRprod[WAVES_PER_BLOCK][NTR][VOX_PER_WAVE + 1];
    __shared__ float sG    [WAVES_PER_BLOCK][NTR][VOX_PER_WAVE + 1];

    const int w    = threadIdx.x >> 5;
    const int lane = threadIdx.x & 31;
    const int voxBase = blockIdx.x * VOX_PER_BLOCK + w * VOX_PER_WAVE;

    // ---- stage geometry: 64 (t,r) x 16 voxels per wave ----
    #pragma unroll 4
    for (int it = 0; it < 32; ++it) {
        const int e   = it * 32 + lane;
        const int tr  = e >> 4;
        const int c   = e & 15;
        const int vox = voxBase + c;
        const int iz = vox & 31, iy = (vox >> 5) & 63, ix = vox >> 11;
        const float px = xc[ix], py = yc[iy], pz = zc[iz];
        const int t = tr >> 3, r = tr & 7;
        const float tx = txp[t*3+0], ty = txp[t*3+1], tz = txp[t*3+2];
        const float rx = rxp[r*3+0], ry = rxp[r*3+1], rz = rxp[r*3+2];
        const float dxt = px - tx, dyt = py - ty, dzt = pz - tz;
        const float dxr = px - rx, dyr = py - ry, dzr = pz - rz;
        const float Rt = sqrtf(dxt*dxt + dyt*dyt + dzt*dzt);
        const float Rr = sqrtf(dxr*dxr + dyr*dyr + dzr*dzr);
        sRsum [w][tr][c] = Rt + Rr;
        sRprod[w][tr][c] = Rt * Rr;
        sG    [w][tr][c] = 4.0f * (dzt / Rt) * (dzr / Rr);
    }
    __syncthreads();

    const float K2PI_C = 6.2831853071795864769f / 299792458.0f;

    v8f accRe[2] = { v8f{}, v8f{} };
    v8f accIm[2] = { v8f{}, v8f{} };

    const int col   = lane & 15;          // voxel column (B / C-D "N")
    const int rhalf = (lane >> 4) << 4;   // B rows 0..7,16..23 (lo) / 8..15,24..31 (hi)

    for (int s = 0; s < KSTEPS; ++s) {
        const float kk  = K2PI_C * freqs[s >> 1];   // uniform -> scalar load
        const float kk2 = kk * kk;
        const int trBase = (s & 1) << 5;

        // prefetch next step's y fragments while we grind sincos
        if (s + 1 < KSTEPS) {
            __builtin_prefetch(&yfre[((s + 1) * 2) * 32 + lane], 0, 1);
            __builtin_prefetch(&yfim[((s + 1) * 2) * 32 + lane], 0, 1);
        }

        // ---- physics B fragments (32x16 bf16, re & im) ----
        v16bf bre, bim;
        #pragma unroll
        for (int v = 0; v < 8; ++v) {
            #pragma unroll
            for (int p = 0; p < 2; ++p) {
                const int row = ((v >> 2) << 4) + (rhalf >> 1) + ((v & 3) << 1) + p;
                const int tr  = trBase + row;
                const float rs = sRsum [w][tr][col];
                const float rp = sRprod[w][tr][col];
                const float gg = sG    [w][tr][col];
                const float ph = kk * rs;
                float sv, cv;
                __sincosf(ph, &sv, &cv);
                const float re0 = gg * (1.0f - kk2 * rp);
                const float im0 = gg * ph;
                bre[2*v + p] = (__bf16)(re0 * cv - im0 * sv);
                bim[2*v + p] = (__bf16)(re0 * sv + im0 * cv);
            }
        }

        // ---- per M-tile: coalesced fragment loads + 4 WMMAs ----
        #pragma unroll
        for (int mt = 0; mt < 2; ++mt) {
            const int fbase = (s * 2 + mt) * 32 + lane;
            const v16bf afr  = yfre[fbase];
            const v16bf afi  = yfim[fbase];
            const v16bf afin = negbf(afi);
            // x_re += y_re*A_re - y_im*A_im ; x_im += y_re*A_im + y_im*A_re
            accRe[mt] = __builtin_amdgcn_wmma_f32_16x16x32_bf16(
                false, afr,  false, bre, (short)0, accRe[mt], false, false);
            accRe[mt] = __builtin_amdgcn_wmma_f32_16x16x32_bf16(
                false, afin, false, bim, (short)0, accRe[mt], false, false);
            accIm[mt] = __builtin_amdgcn_wmma_f32_16x16x32_bf16(
                false, afr,  false, bim, (short)0, accIm[mt], false, false);
            accIm[mt] = __builtin_amdgcn_wmma_f32_16x16x32_bf16(
                false, afi,  false, bre, (short)0, accIm[mt], false, false);
        }
    }

    // ---- epilogue: |x|, coalesced store, wave max + one global atomic ----
    const int mhalf = (lane >> 4) << 3;   // C/D: lanes 0-15 -> M=j, 16-31 -> M=j+8
    float lmax = 0.0f;
    #pragma unroll
    for (int mt = 0; mt < 2; ++mt) {
        #pragma unroll
        for (int j = 0; j < 8; ++j) {
            const float re = accRe[mt][j];
            const float im = accIm[mt][j];
            const float mg = sqrtf(re * re + im * im);
            const int n = mt * 16 + mhalf + j;
            out[(size_t)n * NVOX + voxBase + col] = mg;
            lmax = fmaxf(lmax, mg);
        }
    }
    #pragma unroll
    for (int off = 16; off > 0; off >>= 1)
        lmax = fmaxf(lmax, __shfl_xor(lmax, off, 32));
    if (lane == 0)
        atomicMax(gmax, __float_as_uint(lmax));
}

__global__ void KirchhoffMigration_init_kernel(unsigned int* gmax) {
    *gmax = 0u;
}

__global__ void KirchhoffMigration_norm_kernel(float* __restrict__ out,
                                               const unsigned int* __restrict__ gmax,
                                               int n) {
    const int i = blockIdx.x * blockDim.x + threadIdx.x;
    const float inv = 1.0f / __uint_as_float(*gmax);
    if (i < n) out[i] *= inv;
}

extern "C" void kernel_launch(void* const* d_in, const int* in_sizes, int n_in,
                              void* d_out, int out_size, void* d_ws, size_t ws_size,
                              hipStream_t stream) {
    (void)in_sizes; (void)n_in; (void)ws_size;
    const float* freqs = (const float*)d_in[0];
    const float* txp   = (const float*)d_in[1];
    const float* rxp   = (const float*)d_in[2];
    const float* xc    = (const float*)d_in[3];
    const float* yc    = (const float*)d_in[4];
    const float* zc    = (const float*)d_in[5];
    const float* y_re  = (const float*)d_in[6];
    const float* y_im  = (const float*)d_in[7];
    float* out = (float*)d_out;

    // workspace layout: [0..3] gmax u32; [256 ..] bf16 fragment buffers
    unsigned int* gmax = (unsigned int*)d_ws;
    __bf16* fre = (__bf16*)((char*)d_ws + 256);
    __bf16* fim = fre + FRAG_ELEMS;

    KirchhoffMigration_init_kernel<<<1, 1, 0, stream>>>(gmax);

    KirchhoffMigration_yfrag_kernel<<<(FRAG_ELEMS + 255) / 256, 256, 0, stream>>>(
        y_re, y_im, fre, fim);

    const int blocks = NVOX / VOX_PER_BLOCK;   // 2048
    KirchhoffMigration_22058952032317_kernel<<<blocks, 128, 0, stream>>>(
        freqs, txp, rxp, xc, yc, zc,
        (const v16bf*)fre, (const v16bf*)fim, out, gmax);

    const int total = out_size;                 // 4,194,304
    KirchhoffMigration_norm_kernel<<<(total + 255) / 256, 256, 0, stream>>>(
        out, gmax, total);
}
